// LlamaAttention_38671885533925
// MI455X (gfx1250) — matrix operand order repaired and checked by reading, not compiled, not run on hardware
//
#include <hip/hip_runtime.h>
#include <math.h>

typedef __attribute__((ext_vector_type(16))) __bf16 v16bf;
typedef __attribute__((ext_vector_type(8)))  __bf16 bf16x8;
typedef __attribute__((ext_vector_type(8)))  float  v8f;

#define S_LEN 2048
#define HDIM  2048
#define NH    32
#define NKV   8
#define DH    64
#define KVD   512            // NKV * DH
#define NQKV  3072           // HDIM + 2*KVD  (fused QKV output width)
#define QSCALE 0.125f        // 64^-0.5

// ---------- helpers ----------

__device__ __forceinline__ unsigned short f2bf(float f) {
  unsigned int u = __builtin_bit_cast(unsigned int, f);
  u += 0x7fffu + ((u >> 16) & 1u);   // round-to-nearest-even
  return (unsigned short)(u >> 16);
}

// Build a 16-element bf16 fragment from two contiguous 8-element chunks.
__device__ __forceinline__ v16bf frag16(const unsigned short* lo, const unsigned short* hi) {
  bf16x8 a = *reinterpret_cast<const bf16x8*>(lo);
  bf16x8 b = *reinterpret_cast<const bf16x8*>(hi);
  return __builtin_shufflevector(a, b, 0,1,2,3,4,5,6,7,8,9,10,11,12,13,14,15);
}

__device__ __forceinline__ v16bf ld16(const unsigned short* p) {
  return *reinterpret_cast<const v16bf*>(p);
}

__device__ __forceinline__ v8f wmma_bf16(v16bf a, v16bf b, v8f c) {
  return __builtin_amdgcn_wmma_f32_16x16x32_bf16(false, a, false, b, (short)0, c, false, false);
}

// ---------- elementwise kernels ----------

__global__ void cvt_bf16_k(const float* __restrict__ src, unsigned short* __restrict__ dst, int n) {
  int i = blockIdx.x * blockDim.x + threadIdx.x;
  if (i < n) dst[i] = f2bf(src[i]);
}

// src is [K,N] row-major f32; dst is [N,K] row-major bf16 (i.e. W^T).
__global__ void transpose_bf16_k(const float* __restrict__ src, unsigned short* __restrict__ dst,
                                 int K, int N) {
  int i = blockIdx.x * blockDim.x + threadIdx.x;
  if (i >= N * K) return;
  int n = i / K, k = i - n * K;
  dst[i] = f2bf(src[(size_t)k * N + n]);
}

// RoPE epilogue: q' = (q*cos + rotate_half(q)*sin) * scale, packed to bf16.
// Reads rows of width `width` starting at column `coloff` of a [S, rowstride]
// f32 matrix; writes dense [S, width] bf16. cos/sin are [S,64].
__global__ void rope_bf16_k(const float* __restrict__ qf, const float* __restrict__ cosb,
                            const float* __restrict__ sinb, unsigned short* __restrict__ out,
                            int total, int width, int rowstride, int coloff, float scale) {
  int i = blockIdx.x * blockDim.x + threadIdx.x;
  if (i >= total) return;
  int s = i / width;
  int c = i - s * width;
  int d = c & 63;                 // width % 64 == 0
  size_t src = (size_t)s * rowstride + coloff + c;
  float co = cosb[s * 64 + d], sn = sinb[s * 64 + d];
  float v = qf[src];
  float rot = (d < 32) ? -qf[src + 32] : qf[src - 32];
  out[i] = f2bf((v * co + rot * sn) * scale);
}

// V slice of QKVf [S, NQKV] (cols [coloff, coloff+KVD)) -> Vt [KVD, S] bf16
__global__ void transpose_v_k(const float* __restrict__ qkvf, unsigned short* __restrict__ vt,
                              int coloff) {
  int i = blockIdx.x * blockDim.x + threadIdx.x;
  if (i >= S_LEN * KVD) return;
  int d = i / S_LEN, s = i - d * S_LEN;
  vt[i] = f2bf(qkvf[(size_t)s * NQKV + coloff + d]);
}

// ---------- GEMM: C[M,N] f32 = A[M,K] bf16 @ Bt[N,K]^T bf16 ----------
// One wave computes a 64x64 tile: 4x4 grid of 16x16 accumulators,
// 16 v_wmma_f32_16x16x32_bf16 per K=32 step. Next-tile prefetch via
// global_prefetch_b8 (speculative, dropped if OOB).
__global__ __launch_bounds__(32) void gemm_bf16_k(const unsigned short* __restrict__ A,
                                                  const unsigned short* __restrict__ Bt,
                                                  float* __restrict__ C, int M, int N, int K) {
  const int lane  = threadIdx.x;
  const int lrow  = lane & 15;
  const int lhalf = lane >> 4;
  const int n0 = blockIdx.x * 64;
  const int m0 = blockIdx.y * 64;

  v8f acc[4][4];
#pragma unroll
  for (int i = 0; i < 4; ++i)
#pragma unroll
    for (int j = 0; j < 4; ++j) acc[i][j] = {};

  const unsigned short* arow[4];
  const unsigned short* brow[4];
#pragma unroll
  for (int i = 0; i < 4; ++i) arow[i] = A  + (size_t)(m0 + 16 * i + lrow) * K;
#pragma unroll
  for (int j = 0; j < 4; ++j) brow[j] = Bt + (size_t)(n0 + 16 * j + lrow) * K;

  for (int k0 = 0; k0 < K; k0 += 32) {
    // Pull the k+128 cachelines toward this WGP while the WMMAs run.
#pragma unroll
    for (int i = 0; i < 4; ++i) __builtin_prefetch(arow[i] + k0 + 128, 0, 3);
#pragma unroll
    for (int j = 0; j < 4; ++j) __builtin_prefetch(brow[j] + k0 + 128, 0, 3);

    v16bf af[4], bf[4];
#pragma unroll
    for (int i = 0; i < 4; ++i) {
      const unsigned short* p = arow[i] + k0 + lhalf * 16;    // A layout: 16 contiguous K per lane
      af[i] = frag16(p, p + 8);
    }
#pragma unroll
    for (int j = 0; j < 4; ++j)
      bf[j] = ld16(brow[j] + k0 + lhalf * 16);                // B layout: 16 contiguous K per lane
#pragma unroll
    for (int i = 0; i < 4; ++i)
#pragma unroll
      for (int j = 0; j < 4; ++j)
        acc[i][j] = wmma_bf16(af[i], bf[j], acc[i][j]);
  }

#pragma unroll
  for (int i = 0; i < 4; ++i)
#pragma unroll
    for (int j = 0; j < 4; ++j)
#pragma unroll
      for (int r = 0; r < 8; ++r) {
        int row = m0 + 16 * i + r + lhalf * 8;                // C layout: rows r / r+8 per half
        int col = n0 + 16 * j + lrow;
        C[(size_t)row * N + col] = acc[i][j][r];
      }
}

// ---------- Flash attention: one wave per (16-query tile, head) ----------
// Qb [S, H*D] bf16 (SCALE folded in), Kb [S, KV*D] bf16, Vt [KV*D, S] bf16,
// Attb [S, H*D] bf16.
__global__ __launch_bounds__(32) void attn_fwd_k(const unsigned short* __restrict__ Qb,
                                                 const unsigned short* __restrict__ Kb,
                                                 const unsigned short* __restrict__ Vt,
                                                 unsigned short* __restrict__ Attb) {
  __shared__ unsigned short p_lds[16 * 32];   // P tile, row-major [16 q rows][32 keys]

  const int lane  = threadIdx.x;
  const int lrow  = lane & 15;
  const int lhalf = lane >> 4;
  const int qblk  = blockIdx.x;
  const int h     = blockIdx.y;
  const int g     = h >> 2;                   // kv group = h / REP

  // Q A-fragments for d-chunks [0,32) and [32,64)
  v16bf qa[2];
  {
    const unsigned short* qp = Qb + (size_t)(qblk * 16 + lrow) * HDIM + h * DH;
#pragma unroll
    for (int c = 0; c < 2; ++c) {
      const unsigned short* p = qp + c * 32 + lhalf * 16;
      qa[c] = frag16(p, p + 8);
    }
  }

  float m_s[8], l_s[8];
  v8f o[4];
#pragma unroll
  for (int r = 0; r < 8; ++r) { m_s[r] = -3.0e38f; l_s[r] = 0.0f; }
#pragma unroll
  for (int j = 0; j < 4; ++j) o[j] = {};

  for (int kb = 0; kb < S_LEN; kb += 32) {
    // ---- scores for two 16-key tiles ----
    v8f sc[2];
#pragma unroll
    for (int t = 0; t < 2; ++t) {
      v8f acc = {};
#pragma unroll
      for (int c = 0; c < 2; ++c) {
        const unsigned short* kp =
            Kb + (size_t)(kb + 16 * t + lrow) * KVD + g * DH + c * 32 + lhalf * 16;
        acc = wmma_bf16(qa[c], ld16(kp), acc);
      }
      sc[t] = acc;
    }

    // ---- streaming softmax (rows split across lane halves; reduce width 16) ----
    float mb[8];
#pragma unroll
    for (int r = 0; r < 8; ++r) mb[r] = fmaxf(sc[0][r], sc[1][r]);
#pragma unroll
    for (int off = 1; off < 16; off <<= 1)
#pragma unroll
      for (int r = 0; r < 8; ++r) mb[r] = fmaxf(mb[r], __shfl_xor(mb[r], off, 16));

    float alpha[8], rs[8];
#pragma unroll
    for (int r = 0; r < 8; ++r) {
      float mn = fmaxf(m_s[r], mb[r]);
      alpha[r] = __expf(m_s[r] - mn);
      m_s[r] = mn;
      rs[r] = 0.0f;
    }
#pragma unroll
    for (int t = 0; t < 2; ++t)
#pragma unroll
      for (int r = 0; r < 8; ++r) {
        float p = __expf(sc[t][r] - m_s[r]);
        rs[r] += p;
        p_lds[(r + lhalf * 8) * 32 + t * 16 + lrow] = f2bf(p);
      }
#pragma unroll
    for (int off = 1; off < 16; off <<= 1)
#pragma unroll
      for (int r = 0; r < 8; ++r) rs[r] += __shfl_xor(rs[r], off, 16);
#pragma unroll
    for (int r = 0; r < 8; ++r) l_s[r] = l_s[r] * alpha[r] + rs[r];
#pragma unroll
    for (int j = 0; j < 4; ++j)
#pragma unroll
      for (int r = 0; r < 8; ++r) o[j][r] = o[j][r] * alpha[r];

    __syncthreads();  // single-wave WG: orders the DS stores before the reads

    // ---- P(16x32) @ V(32x64): A-frag from LDS, B-frags contiguous along keys ----
    v16bf pa;
    {
      const unsigned short* pp = p_lds + lrow * 32 + lhalf * 16;
      pa = frag16(pp, pp + 8);
    }
#pragma unroll
    for (int j = 0; j < 4; ++j) {
      const unsigned short* vp =
          Vt + (size_t)(g * DH + 16 * j + lrow) * S_LEN + kb + lhalf * 16;
      o[j] = wmma_bf16(pa, ld16(vp), o[j]);
    }
    __syncthreads();
  }

  // ---- normalize and store bf16 attention output ----
  float inv[8];
#pragma unroll
  for (int r = 0; r < 8; ++r) inv[r] = 1.0f / l_s[r];
#pragma unroll
  for (int j = 0; j < 4; ++j)
#pragma unroll
    for (int r = 0; r < 8; ++r) {
      int row = qblk * 16 + r + lhalf * 8;
      int col = h * DH + 16 * j + lrow;
      Attb[(size_t)row * HDIM + col] = f2bf(o[j][r] * inv[r]);
    }
}

// ---------- host ----------

extern "C" void kernel_launch(void* const* d_in, const int* in_sizes, int n_in,
                              void* d_out, int out_size, void* d_ws, size_t ws_size,
                              hipStream_t stream) {
  (void)in_sizes; (void)n_in; (void)out_size; (void)ws_size;

  const float* x    = (const float*)d_in[0];
  const float* cosb = (const float*)d_in[1];
  const float* sinb = (const float*)d_in[2];
  const float* Wq   = (const float*)d_in[3];
  const float* Wk   = (const float*)d_in[4];
  const float* Wv   = (const float*)d_in[5];
  const float* Wo   = (const float*)d_in[6];

  char* ws = (char*)d_ws;
  size_t off = 0;
  auto carve = [&](size_t bytes) -> char* {
    char* p = ws + off;
    off += (bytes + 255) & ~(size_t)255;
    return p;
  };

  unsigned short* Xb   = (unsigned short*)carve((size_t)S_LEN * HDIM * 2); // x in bf16
  unsigned short* WoT  = (unsigned short*)carve((size_t)HDIM * HDIM * 2);
  unsigned short* WT   = (unsigned short*)carve((size_t)NQKV * HDIM * 2); // [Wq;Wk;Wv]^T fused
  float*          QKVf = (float*)carve((size_t)S_LEN * NQKV * 4);         // fused QKV output
  unsigned short* Kb   = (unsigned short*)carve((size_t)S_LEN * KVD * 2);
  unsigned short* Vtb  = (unsigned short*)carve((size_t)KVD * S_LEN * 2);
  unsigned short* Qb   = (unsigned short*)WT;    // reuse: WT dead after QKV GEMM
  unsigned short* Attb = (unsigned short*)QKVf;  // reuse: QKVf dead after RoPE/V-transpose

  const int TPB = 256;

  // 1) convert x; build fused transposed weight [NQKV, HDIM] = [Wq^T; Wk^T; Wv^T]
  cvt_bf16_k<<<(S_LEN * HDIM) / TPB, TPB, 0, stream>>>(x, Xb, S_LEN * HDIM);
  transpose_bf16_k<<<(HDIM * HDIM) / TPB, TPB, 0, stream>>>(Wq, WT, HDIM, HDIM);
  transpose_bf16_k<<<(HDIM * KVD) / TPB, TPB, 0, stream>>>(
      Wk, WT + (size_t)HDIM * HDIM, HDIM, KVD);
  transpose_bf16_k<<<(HDIM * KVD) / TPB, TPB, 0, stream>>>(
      Wv, WT + (size_t)(HDIM + KVD) * HDIM, HDIM, KVD);
  transpose_bf16_k<<<(HDIM * HDIM) / TPB, TPB, 0, stream>>>(Wo, WoT, HDIM, HDIM);

  // 2) fused QKV projection: [S, HDIM] @ [HDIM, NQKV] -> [S, NQKV] f32
  gemm_bf16_k<<<dim3(NQKV / 64, S_LEN / 64), 32, 0, stream>>>(Xb, WT, QKVf,
                                                              S_LEN, NQKV, HDIM);

  // 3) RoPE (+fold softmax scale into Q), pack to bf16; transpose V
  rope_bf16_k<<<(S_LEN * HDIM) / TPB, TPB, 0, stream>>>(
      QKVf, cosb, sinb, Qb, S_LEN * HDIM, HDIM, NQKV, 0, QSCALE);
  rope_bf16_k<<<(S_LEN * KVD) / TPB, TPB, 0, stream>>>(
      QKVf, cosb, sinb, Kb, S_LEN * KVD, KVD, NQKV, HDIM, 1.0f);
  transpose_v_k<<<(S_LEN * KVD) / TPB, TPB, 0, stream>>>(QKVf, Vtb, HDIM + KVD);

  // 4) flash attention: one wave per (16 queries x head)
  attn_fwd_k<<<dim3(S_LEN / 16, NH), 32, 0, stream>>>(Qb, Kb, Vtb, Attb);

  // 5) output projection straight to d_out (f32)
  gemm_bf16_k<<<dim3(HDIM / 64, S_LEN / 64), 32, 0, stream>>>(Attb, WoT, (float*)d_out,
                                                              S_LEN, HDIM, HDIM);
}
